// GraphEncoder_51891794871088
// MI455X (gfx1250) — matrix operand order repaired and checked by reading, not compiled, not run on hardware
//
#include <hip/hip_runtime.h>

// ---------------------------------------------------------------------------
// GraphNets encoder for MI455X (gfx1250, wave32, WMMA 16x16x32 f16->f32).
//
// Pipeline (all on `stream`):
//   1. zero_kernel      : clear agg[N,128] + v_sum[128] + e_sum[128] in ws
//   2. swizzle_kernel x4: f32 weights -> f16, pre-swizzled into WMMA B-fragment
//                         layout (lane-contiguous 16-half fragments)
//   3. edge_mlp_kernel  : 64 edges/block, gather->LDS f16, 2x WMMA GEMM layers,
//                         write e, scatter-add to agg, reduce e_sum
//   4. node_mlp_kernel  : 64 nodes/block, same structure, write v, reduce v_sum
//   5. global_mlp_kernel: means + tiny 288->128->128 MLP, write g
// ---------------------------------------------------------------------------

typedef __attribute__((ext_vector_type(16))) _Float16 v16h;
typedef __attribute__((ext_vector_type(8)))  _Float16 v8h;
typedef __attribute__((ext_vector_type(8)))  float    v8f;

static constexpr int HD  = 128;  // hidden width
static constexpr int DV  = 64;   // node feature dim
static constexpr int DE  = 32;   // edge feature dim
static constexpr int DU  = 32;   // global feature dim
static constexpr int KE  = 2 * DV + DE + DU; // 192, edge MLP input dim
static constexpr int KN  = DV + HD + DU;     // 224, node MLP input dim
static constexpr int KTE = KE / 32;  // 6  K-tiles, edge layer 1
static constexpr int KTN = KN / 32;  // 7  K-tiles, node layer 1
static constexpr int KTH = HD / 32;  // 4  K-tiles, layer 2
static constexpr int FRAG_PER_KT = 8 * 32 * 16; // 4096 halves per K-tile group

// ---- WMMA helpers ---------------------------------------------------------

__device__ __forceinline__ v8f wmma_f16(v16h a, v16h b, v8f c) {
  // (neg_a, A, neg_b, B, c_mod, C, reuse_a, reuse_b)
  return __builtin_amdgcn_wmma_f32_16x16x32_f16(false, a, false, b,
                                                (short)0, c, false, false);
}

// A-fragment (16x32 f16) from a row-major LDS tile, per ISA 7.12.2 layout:
//   lanes 0-15 : M=lane,    halves 0..7 = K kt*32+0..7,  halves 8..15 = +16..23
//   lanes 16-31: M=lane-16, halves 0..7 = K kt*32+8..15, halves 8..15 = +24..31
__device__ __forceinline__ v16h load_a_frag(const _Float16* base, int row,
                                            int kt, int lane, int kstride) {
  const int kb = kt * 32 + ((lane & 16) ? 8 : 0);
  const _Float16* p = base + row * kstride + kb;
  v8h lo = *(const v8h*)(p);
  v8h hi = *(const v8h*)(p + 16);
  v16h a;
#pragma unroll
  for (int i = 0; i < 8; ++i) { a[i] = lo[i]; a[i + 8] = hi[i]; }
  return a;
}

// ---- small utility kernels ------------------------------------------------

__global__ void zero_kernel(float* __restrict__ p, long long n) {
  long long i = (long long)blockIdx.x * 256 + threadIdx.x;
  if (i < n) p[i] = 0.0f;
}

// Pre-swizzle an f32 weight matrix W[K x 128] into f16 WMMA B fragments:
// out[((kt*8+nt)*32+lane)*16 + j] = W[kt*32 + (lane>=16?8:0) + (j&7) + (j>=8?16:0)][nt*16 + (lane&15)]
__global__ void swizzle_kernel(const float* __restrict__ W,
                               _Float16* __restrict__ out, int KT) {
  const int idx = blockIdx.x * 256 + threadIdx.x;
  if (idx >= KT * FRAG_PER_KT) return;
  const int j    = idx & 15;
  const int lane = (idx >> 4) & 31;
  const int nt   = (idx >> 9) & 7;
  const int kt   = idx >> 12;
  const int k = kt * 32 + ((lane >> 4) << 3) + (j & 7) + ((j >> 3) << 4);
  const int n = nt * 16 + (lane & 15);
  out[idx] = (_Float16)W[k * HD + n];
}

// ---- edge MLP: e = relu(e_in @ W1 + b1) @ W2 + b2 -------------------------

__global__ void __launch_bounds__(256) edge_mlp_kernel(
    const float* __restrict__ node_attr, const int* __restrict__ edges,
    const float* __restrict__ edge_attr, const float* __restrict__ u,
    const _Float16* __restrict__ w1, const float* __restrict__ b1,
    const _Float16* __restrict__ w2, const float* __restrict__ b2,
    float* __restrict__ e_out, float* __restrict__ agg,
    float* __restrict__ e_sum, int Nn, long long Ee) {
  __shared__ _Float16 s_in[64 * KE];   // 24.0 KB
  __shared__ _Float16 s_hid[64 * HD];  // 16.0 KB
  __shared__ float    s_sum[HD];       //  0.5 KB

  const int tid  = threadIdx.x;
  const int lane = tid & 31;
  const int wave = tid >> 5;
  const long long e0 = (long long)blockIdx.x * 64;
  const int* rowi = edges;
  const int* coli = edges + Ee;

  if (tid < HD) s_sum[tid] = 0.0f;
  if (tid < 48) {  // warm L2 for the streamed f16 weight fragments
    __builtin_prefetch(w1 + tid * 512, 0, 3);
    __builtin_prefetch(w2 + tid * 256, 0, 3);
  }

  // Gather + f16 convert: 64 edges x 192 features -> LDS row-major
  for (int i = tid; i < 64 * KE; i += 256) {
    const int el = i / KE;
    const int d  = i - el * KE;
    long long eg = e0 + el;
    if (eg >= Ee) eg = Ee - 1;
    float x;
    if (d < DV)                 x = node_attr[(long long)rowi[eg] * DV + d];
    else if (d < 2 * DV)        x = node_attr[(long long)coli[eg] * DV + (d - DV)];
    else if (d < 2 * DV + DE)   x = edge_attr[eg * DE + (d - 2 * DV)];
    else                        x = u[d - (2 * DV + DE)];
    s_in[i] = (_Float16)x;
  }
  __syncthreads();

  // Layer 1: [64x192]f16 x [192x128]f16 + b1, ReLU -> s_hid (f16)
#pragma unroll
  for (int t = 0; t < 4; ++t) {
    const int tile = wave * 4 + t;
    const int mt = tile >> 3, nt = tile & 7;
    const int n = nt * 16 + (lane & 15);
    v8f acc;
    const float bv = b1[n];
#pragma unroll
    for (int i = 0; i < 8; ++i) acc[i] = bv;
#pragma unroll
    for (int kt = 0; kt < KTE; ++kt) {
      v16h a = load_a_frag(s_in, mt * 16 + (lane & 15), kt, lane, KE);
      v16h b = *(const v16h*)(w1 + ((size_t)(kt * 8 + nt) * 32 + lane) * 16);
      acc = wmma_f16(a, b, acc);
    }
#pragma unroll
    for (int i = 0; i < 8; ++i) {
      const float x = acc[i] > 0.0f ? acc[i] : 0.0f;
      const int m = ((lane >> 4) << 3) + i;
      s_hid[(mt * 16 + m) * HD + n] = (_Float16)x;
    }
  }
  __syncthreads();

  // Layer 2: [64x128]f16 x [128x128]f16 + b2 -> e, scatter-add agg, e_sum
#pragma unroll
  for (int t = 0; t < 4; ++t) {
    const int tile = wave * 4 + t;
    const int mt = tile >> 3, nt = tile & 7;
    const int n = nt * 16 + (lane & 15);
    v8f acc;
    const float bv = b2[n];
#pragma unroll
    for (int i = 0; i < 8; ++i) acc[i] = bv;
#pragma unroll
    for (int kt = 0; kt < KTH; ++kt) {
      v16h a = load_a_frag(s_hid, mt * 16 + (lane & 15), kt, lane, HD);
      v16h b = *(const v16h*)(w2 + ((size_t)(kt * 8 + nt) * 32 + lane) * 16);
      acc = wmma_f16(a, b, acc);
    }
    float psum = 0.0f;
#pragma unroll
    for (int i = 0; i < 8; ++i) {
      const int m = ((lane >> 4) << 3) + i;
      const long long eg = e0 + mt * 16 + m;
      if (eg < Ee) {
        const float x = acc[i];
        e_out[eg * HD + n] = x;
        __hip_atomic_fetch_add(&agg[(long long)coli[eg] * HD + n], x,
                               __ATOMIC_RELAXED, __HIP_MEMORY_SCOPE_AGENT);
        psum += x;
      }
    }
    __hip_atomic_fetch_add(&s_sum[n], psum, __ATOMIC_RELAXED,
                           __HIP_MEMORY_SCOPE_WORKGROUP);
  }
  __syncthreads();
  if (tid < HD)
    __hip_atomic_fetch_add(&e_sum[tid], s_sum[tid], __ATOMIC_RELAXED,
                           __HIP_MEMORY_SCOPE_AGENT);
}

// ---- node MLP: v = relu(v_in @ W1 + b1) @ W2 + b2 -------------------------

__global__ void __launch_bounds__(256) node_mlp_kernel(
    const float* __restrict__ node_attr, const float* __restrict__ agg,
    const float* __restrict__ u,
    const _Float16* __restrict__ w1, const float* __restrict__ b1,
    const _Float16* __restrict__ w2, const float* __restrict__ b2,
    float* __restrict__ v_out, float* __restrict__ v_sum, int Nn) {
  __shared__ _Float16 s_in[64 * KN];   // 28.0 KB
  __shared__ _Float16 s_hid[64 * HD];  // 16.0 KB
  __shared__ float    s_sum[HD];

  const int tid  = threadIdx.x;
  const int lane = tid & 31;
  const int wave = tid >> 5;
  const long long n0 = (long long)blockIdx.x * 64;

  if (tid < HD) s_sum[tid] = 0.0f;
  if (tid < 56) {
    __builtin_prefetch(w1 + tid * 512, 0, 3);
    __builtin_prefetch(w2 + tid * 256, 0, 3);
  }

  // Gather + f16 convert: 64 nodes x 224 features -> LDS row-major
  for (int i = tid; i < 64 * KN; i += 256) {
    const int el = i / KN;
    const int d  = i - el * KN;
    long long nid = n0 + el;
    if (nid >= Nn) nid = Nn - 1;
    float x;
    if (d < DV)            x = node_attr[nid * DV + d];
    else if (d < DV + HD)  x = agg[nid * HD + (d - DV)];
    else                   x = u[d - (DV + HD)];
    s_in[i] = (_Float16)x;
  }
  __syncthreads();

  // Layer 1: [64x224] x [224x128] + b1, ReLU -> s_hid
#pragma unroll
  for (int t = 0; t < 4; ++t) {
    const int tile = wave * 4 + t;
    const int mt = tile >> 3, nt = tile & 7;
    const int n = nt * 16 + (lane & 15);
    v8f acc;
    const float bv = b1[n];
#pragma unroll
    for (int i = 0; i < 8; ++i) acc[i] = bv;
#pragma unroll
    for (int kt = 0; kt < KTN; ++kt) {
      v16h a = load_a_frag(s_in, mt * 16 + (lane & 15), kt, lane, KN);
      v16h b = *(const v16h*)(w1 + ((size_t)(kt * 8 + nt) * 32 + lane) * 16);
      acc = wmma_f16(a, b, acc);
    }
#pragma unroll
    for (int i = 0; i < 8; ++i) {
      const float x = acc[i] > 0.0f ? acc[i] : 0.0f;
      const int m = ((lane >> 4) << 3) + i;
      s_hid[(mt * 16 + m) * HD + n] = (_Float16)x;
    }
  }
  __syncthreads();

  // Layer 2 -> v, v_sum
#pragma unroll
  for (int t = 0; t < 4; ++t) {
    const int tile = wave * 4 + t;
    const int mt = tile >> 3, nt = tile & 7;
    const int n = nt * 16 + (lane & 15);
    v8f acc;
    const float bv = b2[n];
#pragma unroll
    for (int i = 0; i < 8; ++i) acc[i] = bv;
#pragma unroll
    for (int kt = 0; kt < KTH; ++kt) {
      v16h a = load_a_frag(s_hid, mt * 16 + (lane & 15), kt, lane, HD);
      v16h b = *(const v16h*)(w2 + ((size_t)(kt * 8 + nt) * 32 + lane) * 16);
      acc = wmma_f16(a, b, acc);
    }
    float psum = 0.0f;
#pragma unroll
    for (int i = 0; i < 8; ++i) {
      const int m = ((lane >> 4) << 3) + i;
      const long long nid = n0 + mt * 16 + m;
      if (nid < Nn) {
        const float x = acc[i];
        v_out[nid * HD + n] = x;
        psum += x;
      }
    }
    __hip_atomic_fetch_add(&s_sum[n], psum, __ATOMIC_RELAXED,
                           __HIP_MEMORY_SCOPE_WORKGROUP);
  }
  __syncthreads();
  if (tid < HD)
    __hip_atomic_fetch_add(&v_sum[tid], s_sum[tid], __ATOMIC_RELAXED,
                           __HIP_MEMORY_SCOPE_AGENT);
}

// ---- global MLP: g = relu(concat(nv, ne, u) @ W1 + b1) @ W2 + b2 ----------

__global__ void global_mlp_kernel(const float* __restrict__ v_sum,
                                  const float* __restrict__ e_sum,
                                  const float* __restrict__ u,
                                  const float* __restrict__ W1,
                                  const float* __restrict__ b1,
                                  const float* __restrict__ W2,
                                  const float* __restrict__ b2,
                                  float* __restrict__ g_out,
                                  int Nn, long long Ee) {
  __shared__ float gin[2 * HD + DU];
  __shared__ float hid[HD];
  const int t = threadIdx.x;  // 128 threads
  gin[t]      = v_sum[t] / (float)Nn;
  gin[HD + t] = e_sum[t] / (float)Ee;
  if (t < DU) gin[2 * HD + t] = u[t];
  __syncthreads();
  float a = b1[t];
  for (int k = 0; k < 2 * HD + DU; ++k) a += gin[k] * W1[k * HD + t];
  hid[t] = a > 0.0f ? a : 0.0f;
  __syncthreads();
  float o = b2[t];
  for (int k = 0; k < HD; ++k) o += hid[k] * W2[k * HD + t];
  g_out[t] = o;
}

// ---------------------------------------------------------------------------

extern "C" void kernel_launch(void* const* d_in, const int* in_sizes, int n_in,
                              void* d_out, int out_size, void* d_ws,
                              size_t ws_size, hipStream_t stream) {
  const float* node_attr = (const float*)d_in[0];
  const int*   edges     = (const int*)d_in[1];
  const float* edge_attr = (const float*)d_in[2];
  const float* u         = (const float*)d_in[3];
  const int       Nn = in_sizes[4];            // node_idx length = N
  const long long Ee = (long long)in_sizes[5]; // edge_idx length = E
  const float* W1e = (const float*)d_in[6];  const float* b1e = (const float*)d_in[7];
  const float* W2e = (const float*)d_in[8];  const float* b2e = (const float*)d_in[9];
  const float* W1n = (const float*)d_in[10]; const float* b1n = (const float*)d_in[11];
  const float* W2n = (const float*)d_in[12]; const float* b2n = (const float*)d_in[13];
  const float* W1g = (const float*)d_in[14]; const float* b1g = (const float*)d_in[15];
  const float* W2g = (const float*)d_in[16]; const float* b2g = (const float*)d_in[17];

  // Output: concat(v [N,128], e [E,128], g [1,128]) flat
  float* v_out = (float*)d_out;
  float* e_out = v_out + (long long)Nn * HD;
  float* g_out = e_out + Ee * HD;

  // Workspace: agg[N,128] f32 | v_sum[128] | e_sum[128] | f16 weights
  float* agg  = (float*)d_ws;
  float* vsum = agg + (long long)Nn * HD;
  float* esum = vsum + HD;
  _Float16* w1e = (_Float16*)(esum + HD);
  _Float16* w2e = w1e + (size_t)KTE * FRAG_PER_KT;
  _Float16* w1n = w2e + (size_t)KTH * FRAG_PER_KT;
  _Float16* w2n = w1n + (size_t)KTN * FRAG_PER_KT;

  const long long nz = (long long)Nn * HD + 2 * HD;
  zero_kernel<<<(int)((nz + 255) / 256), 256, 0, stream>>>(agg, nz);
  swizzle_kernel<<<(KTE * FRAG_PER_KT + 255) / 256, 256, 0, stream>>>(W1e, w1e, KTE);
  swizzle_kernel<<<(KTH * FRAG_PER_KT + 255) / 256, 256, 0, stream>>>(W2e, w2e, KTH);
  swizzle_kernel<<<(KTN * FRAG_PER_KT + 255) / 256, 256, 0, stream>>>(W1n, w1n, KTN);
  swizzle_kernel<<<(KTH * FRAG_PER_KT + 255) / 256, 256, 0, stream>>>(W2n, w2n, KTH);

  const int eblocks = (int)((Ee + 63) / 64);
  edge_mlp_kernel<<<eblocks, 256, 0, stream>>>(node_attr, edges, edge_attr, u,
                                               w1e, b1e, w2e, b2e, e_out, agg,
                                               esum, Nn, Ee);
  const int nblocks = (Nn + 63) / 64;
  node_mlp_kernel<<<nblocks, 256, 0, stream>>>(node_attr, agg, u, w1n, b1n,
                                               w2n, b2n, v_out, vsum, Nn);
  global_mlp_kernel<<<1, HD, 0, stream>>>(vsum, esum, u, W1g, b1g, W2g, b2g,
                                          g_out, Nn, Ee);
}
